// CondVQModule_76347338654344
// MI455X (gfx1250) — matrix operand order repaired and checked by reading, not compiled, not run on hardware
//
#include <hip/hip_runtime.h>
#include <hip/hip_bf16.h>
#include <math.h>

// ---------------------------------------------------------------------------
// CondVQ transformer forward for MI455X (gfx1250, wave32, WMMA).
// f16 activation/weight pipeline, V_WMMA_F32_16X16X32_F16 everywhere.
// GEMM: 2x2 register-tiled (32x32 C per wave, 4 wmmas per k-step).
// Attention: flash loop over 32 keys/iter, dense PV fragments.
// ---------------------------------------------------------------------------

typedef __attribute__((ext_vector_type(16))) _Float16 v16h;
typedef __attribute__((ext_vector_type(8)))  _Float16 v8h;
typedef __attribute__((ext_vector_type(8)))  float    v8f;

#define B_DIM  32
#define T_DIM  512
#define P_DIM  256
#define H_DIM  256
#define NHEADS 8
#define DH     32
#define CD_DIM 128
#define K_CB   4
#define S_CB   128
#define SUB_D  32
#define L_DIM  256
#define FRAME_K  403
#define FRAME_KP 416   // padded to multiple of 32

// d_out flat layout (floats)
#define POSE_N  (B_DIM*T_DIM*144)
#define TRANS_N (B_DIM*T_DIM*3)
#define PTS_N   (B_DIM*P_DIM*3)
#define IDXS_N  (B_DIM*L_DIM*K_CB)
#define TRANS_OFF (POSE_N)
#define PTS_OFF   (POSE_N+TRANS_N)
#define IDX_OFF   (POSE_N+TRANS_N+PTS_N)
#define SC_OFF    (POSE_N+TRANS_N+PTS_N+IDXS_N)

__device__ __forceinline__ float gelu_exact(float x) {
  return 0.5f * x * (1.0f + erff(x * 0.70710678118654752f));
}

__device__ __forceinline__ v16h cat8(v8h lo, v8h hi) {
  return __builtin_shufflevector(lo, hi, 0, 1, 2, 3, 4, 5, 6, 7,
                                 8, 9, 10, 11, 12, 13, 14, 15);
}

#define WMMA_F16(a, b, c) \
  __builtin_amdgcn_wmma_f32_16x16x32_f16(false, (a), false, (b), (short)0, (c), false, false)

// ---------------------------------------------------------------------------
// GEMM: C[M,N] = act(A[M,K] @ W[N,K]^T + bias[N]); A,W f16; K % 32 == 0.
// Wave computes a 32x32 C tile (2x2 fragments); block = 8 waves (4M x 2N)
// covering 128x64. A fragment = two b128 loads at k0+half*8 / +16;
// B fragment = two b128 loads at k0+half*8 / +16. All loads unconditional
// (N-partial tiles clamp the row pointer; masking done at the store).
// ---------------------------------------------------------------------------
__global__ void __launch_bounds__(256)
gemm_f16(const _Float16* __restrict__ A, const _Float16* __restrict__ Wt,
         const float* __restrict__ bias, void* __restrict__ C,
         int M, int N, int K, int act, int c_f32)
{
  const int wave = threadIdx.x >> 5;
  const int lane = threadIdx.x & 31;
  const int half = lane >> 4;
  const int l16  = lane & 15;
  const int wm = wave & 3, wn = wave >> 2;
  const int m0 = blockIdx.x * 128 + wm * 32;
  const int n0 = blockIdx.y * 64 + wn * 32;
  if (m0 >= M) return;                       // wave-uniform exit

  v8f acc00 = {}, acc01 = {}, acc10 = {}, acc11 = {};
  const _Float16* Ar0 = A + (size_t)(m0 + l16) * K + half * 8;
  const _Float16* Ar1 = A + (size_t)(m0 + 16 + l16) * K + half * 8;
  const int bn0 = n0 + l16, bn1 = n0 + 16 + l16;
  const _Float16* Wr0 = Wt + (size_t)((bn0 < N) ? bn0 : 0) * K + half * 8;
  const _Float16* Wr1 = Wt + (size_t)((bn1 < N) ? bn1 : 0) * K + half * 8;

  for (int k0 = 0; k0 < K; k0 += 32) {
    v16h a0 = cat8(*(const v8h*)(Ar0 + k0), *(const v8h*)(Ar0 + k0 + 16));
    v16h a1 = cat8(*(const v8h*)(Ar1 + k0), *(const v8h*)(Ar1 + k0 + 16));
    v16h b0 = cat8(*(const v8h*)(Wr0 + k0), *(const v8h*)(Wr0 + k0 + 16));
    v16h b1 = cat8(*(const v8h*)(Wr1 + k0), *(const v8h*)(Wr1 + k0 + 16));
    acc00 = WMMA_F16(a0, b0, acc00);
    acc01 = WMMA_F16(a0, b1, acc01);
    acc10 = WMMA_F16(a1, b0, acc10);
    acc11 = WMMA_F16(a1, b1, acc11);
  }

  auto emit = [&](int m, int n, float v) {
    if (m < M && n < N) {
      v += bias[n];
      if (act == 1) v = gelu_exact(v);
      if (c_f32) ((float*)C)[(size_t)m * N + n] = v;
      else       ((_Float16*)C)[(size_t)m * N + n] = (_Float16)v;
    }
  };
#pragma unroll
  for (int r = 0; r < 8; ++r) {
    int mA = m0 + r + half * 8, mB = mA + 16;
    int nA = n0 + l16,          nB = nA + 16;
    emit(mA, nA, acc00[r]);
    emit(mA, nB, acc01[r]);
    emit(mB, nA, acc10[r]);
    emit(mB, nB, acc11[r]);
  }
}

// ---------------------------------------------------------------------------
// Flash attention, one wave per (b, head, 16-query tile); 32 keys per
// iteration: 2 score wmmas + dense 16x32 P tile (via LDS) + 2 PV wmmas.
// qkv is f16 (B,T,3,NHEADS,DH) contiguous; row stride 768 halves.
// ---------------------------------------------------------------------------
__global__ void __launch_bounds__(32)
attn_flash(const _Float16* __restrict__ qkv, _Float16* __restrict__ ctx, int T)
{
  __shared__ _Float16 pbuf[16 * 32];
  const int lane = threadIdx.x & 31;
  const int half = lane >> 4;
  const int l16  = lane & 15;
  const int nqt  = T >> 4;
  int bid = blockIdx.x;
  const int qt = bid % nqt; bid /= nqt;
  const int hh = bid % NHEADS; bid /= NHEADS;
  const int b  = bid;
  const _Float16* base = qkv + (size_t)b * T * (3 * H_DIM);

  // Q fragment (A layout), 1/sqrt(dh) folded in as packed f16 mul
  v16h aq;
  {
    const _Float16* qrow = base + (size_t)(qt * 16 + l16) * (3 * H_DIM)
                         + hh * DH + half * 8;
    const _Float16 hs = (_Float16)0.17677669529663688f;
    aq = cat8((*(const v8h*)(qrow)) * hs, (*(const v8h*)(qrow + 16)) * hs);
  }

  float rowm[8], rowl[8];
  v8f o0 = {}, o1 = {};
#pragma unroll
  for (int r = 0; r < 8; ++r) { rowm[r] = -3.0e38f; rowl[r] = 0.0f; }

  for (int kt2 = 0; kt2 < (T >> 5); ++kt2) {
    const int tb = kt2 * 32;
    // two K fragments (B layout): lane -> key token, dh halves half*8+[0,8) and 16+half*8+[0,8)
    v16h bk0, bk1;
    {
      const _Float16* kr0 = base + (size_t)(tb + l16) * (3 * H_DIM)
                          + H_DIM + hh * DH + half * 8;
      const _Float16* kr1 = kr0 + (size_t)16 * (3 * H_DIM);
      bk0 = cat8(*(const v8h*)(kr0), *(const v8h*)(kr0 + 16));
      bk1 = cat8(*(const v8h*)(kr1), *(const v8h*)(kr1 + 16));
    }
    v8f s0 = {}, s1 = {};
    s0 = WMMA_F16(aq, bk0, s0);
    s1 = WMMA_F16(aq, bk1, s1);

    float p0[8], p1[8];
#pragma unroll
    for (int r = 0; r < 8; ++r) {
      float t0 = s0[r], t1 = s1[r];
      float mx = fmaxf(t0, t1);
#pragma unroll
      for (int msk = 1; msk < 16; msk <<= 1)
        mx = fmaxf(mx, __shfl_xor(mx, msk, 32));     // reduce within half
      float newm = fmaxf(rowm[r], mx);
      float corr = __expf(rowm[r] - newm);
      float e0 = __expf(t0 - newm);
      float e1 = __expf(t1 - newm);
      float ps = e0 + e1;
#pragma unroll
      for (int msk = 1; msk < 16; msk <<= 1)
        ps += __shfl_xor(ps, msk, 32);
      rowl[r] = rowl[r] * corr + ps;
      rowm[r] = newm;
      o0[r] *= corr; o1[r] *= corr;
      p0[r] = e0; p1[r] = e1;
    }
    __syncthreads();
#pragma unroll
    for (int r = 0; r < 8; ++r) {
      pbuf[(r + half * 8) * 32 + l16]      = (_Float16)p0[r];
      pbuf[(r + half * 8) * 32 + 16 + l16] = (_Float16)p1[r];
    }
    __syncthreads();

    // P as full 16x32 A-fragment: two aligned b128 LDS loads
    const _Float16* prow = pbuf + l16 * 32 + half * 8;
    v16h ap = cat8(*(const v8h*)(prow), *(const v8h*)(prow + 16));

    // V fragments: dense, unconditional (element e -> token tb+half*8+e (e<8), tb+16+half*8+(e-8) (e>=8))
    v16h bv0, bv1;
    {
      const _Float16* vbase = base + 2 * H_DIM + hh * DH
                            + (size_t)(tb + half * 8) * (3 * H_DIM);
#pragma unroll
      for (int e = 0; e < 16; ++e) {
        const _Float16* vrow = vbase + (size_t)(e < 8 ? e : e + 8) * (3 * H_DIM);
        bv0[e] = vrow[l16];
        bv1[e] = vrow[16 + l16];
      }
    }
    o0 = WMMA_F16(ap, bv0, o0);
    o1 = WMMA_F16(ap, bv1, o1);
  }
#pragma unroll
  for (int r = 0; r < 8; ++r) {
    int t = qt * 16 + r + half * 8;
    float inv = 1.0f / rowl[r];
    _Float16* crow = ctx + ((size_t)b * T + t) * H_DIM + hh * DH;
    crow[l16]      = (_Float16)(o0[r] * inv);
    crow[16 + l16] = (_Float16)(o1[r] * inv);
  }
}

// ---------------------------------------------------------------------------
// LayerNorm(x + y), H=256, one wave per row; f16 in/out, f32 stats.
// ---------------------------------------------------------------------------
__global__ void __launch_bounds__(256)
ln_residual(const _Float16* __restrict__ x, const _Float16* __restrict__ y,
            const float* __restrict__ w, const float* __restrict__ bb,
            _Float16* __restrict__ out, int rows)
{
  const int wave = threadIdx.x >> 5;
  const int lane = threadIdx.x & 31;
  const int row  = blockIdx.x * 8 + wave;
  if (row >= rows) return;
  const _Float16* xr = x + (size_t)row * H_DIM;
  const _Float16* yr = y + (size_t)row * H_DIM;
  float v[8];
  float s = 0.f;
#pragma unroll
  for (int i = 0; i < 8; ++i) {
    v[i] = (float)xr[lane + i * 32] + (float)yr[lane + i * 32];
    s += v[i];
  }
#pragma unroll
  for (int msk = 1; msk < 32; msk <<= 1) s += __shfl_xor(s, msk, 32);
  float mu = s * (1.0f / 256.0f);
  float vs = 0.f;
#pragma unroll
  for (int i = 0; i < 8; ++i) { float d = v[i] - mu; vs += d * d; }
#pragma unroll
  for (int msk = 1; msk < 32; msk <<= 1) vs += __shfl_xor(vs, msk, 32);
  float rinv = rsqrtf(vs * (1.0f / 256.0f) + 1e-5f);
  _Float16* orow = out + (size_t)row * H_DIM;
#pragma unroll
  for (int i = 0; i < 8; ++i) {
    int e = lane + i * 32;
    orow[e] = (_Float16)((v[i] - mu) * rinv * w[e] + bb[e]);
  }
}

// ---------------------------------------------------------------------------
// Conversions / reshaping
// ---------------------------------------------------------------------------
__global__ void cvt_f32_f16(const float* __restrict__ src,
                            _Float16* __restrict__ dst, int n)
{
  int i = blockIdx.x * blockDim.x + threadIdx.x;
  if (i < n) dst[i] = (_Float16)src[i];
}

// rows x K (f32) -> rows x Kp (f16), zero padded
__global__ void cvt_pad_f32_f16(const float* __restrict__ src,
                                _Float16* __restrict__ dst,
                                int rows, int K, int Kp)
{
  int i = blockIdx.x * blockDim.x + threadIdx.x;
  if (i >= rows * Kp) return;
  int k = i % Kp, r = i / Kp;
  dst[i] = (k < K) ? (_Float16)src[(size_t)r * K + k] : (_Float16)0.0f;
}

__global__ void concat_frame(const float* __restrict__ pose,
                             const float* __restrict__ trans,
                             const float* __restrict__ pt,
                             _Float16* __restrict__ frame, int total)
{
  int i = blockIdx.x * blockDim.x + threadIdx.x;
  if (i >= total) return;
  int col = i % FRAME_KP;
  int row = i / FRAME_KP;            // b*T + t
  int b = row / T_DIM;
  float v;
  if (col < 144)             v = pose[(size_t)row * 144 + col];
  else if (col < 147)        v = trans[(size_t)row * 3 + (col - 144)];
  else if (col < FRAME_K)    v = pt[(size_t)b * H_DIM + (col - 147)];
  else                       v = 0.0f;
  frame[i] = (_Float16)v;
}

__global__ void mean_h2f(const _Float16* __restrict__ in, float* __restrict__ out,
                         int B, int R, int C)
{
  int i = blockIdx.x * blockDim.x + threadIdx.x;
  if (i >= B * C) return;
  int b = i / C, c = i % C;
  float s = 0.f;
  for (int r = 0; r < R; ++r) s += (float)in[((size_t)b * R + r) * C + c];
  out[i] = s / (float)R;
}

__global__ void mean_h2h(const _Float16* __restrict__ in, _Float16* __restrict__ out,
                         int B, int R, int C)
{
  int i = blockIdx.x * blockDim.x + threadIdx.x;
  if (i >= B * C) return;
  int b = i / C, c = i % C;
  float s = 0.f;
  for (int r = 0; r < R; ++r) s += (float)in[((size_t)b * R + r) * C + c];
  out[i] = (_Float16)(s / (float)R);
}

__global__ void take_stride2(const _Float16* __restrict__ x,
                             _Float16* __restrict__ out, int total)
{
  int i = blockIdx.x * blockDim.x + threadIdx.x;
  if (i >= total) return;
  int c = i % H_DIM; int row = i / H_DIM;
  int b = row / L_DIM, l = row % L_DIM;
  out[i] = x[((size_t)b * T_DIM + 2 * l) * H_DIM + c];
}

__global__ void repeat2(const _Float16* __restrict__ h8,
                        _Float16* __restrict__ out, int total)
{
  int i = blockIdx.x * blockDim.x + threadIdx.x;
  if (i >= total) return;
  int c = i % H_DIM; int row = i / H_DIM;
  int b = row / T_DIM, t = row % T_DIM;
  out[i] = h8[((size_t)b * L_DIM + (t >> 1)) * H_DIM + c];
}

// ---------------------------------------------------------------------------
// Product quantization (z_e f32, codebooks f32; z_q written f16)
// ---------------------------------------------------------------------------
__global__ void quantize_kernel(const float* __restrict__ ze,
                                const float* __restrict__ cb,
                                _Float16* __restrict__ zq,
                                float* __restrict__ idx_out,
                                int* __restrict__ hist,
                                float* __restrict__ sqerr)
{
  int gid = blockIdx.x * blockDim.x + threadIdx.x;
  if (gid >= B_DIM * L_DIM * K_CB) return;
  int k = gid % K_CB;
  int row = gid / K_CB;
  const float* z = ze + (size_t)row * CD_DIM + k * SUB_D;
  float zl[SUB_D];
#pragma unroll
  for (int d = 0; d < SUB_D; ++d) zl[d] = z[d];
  int best = 0; float bestd = 3.0e38f;
  for (int s = 0; s < S_CB; ++s) {
    const float* c = cb + ((size_t)k * S_CB + s) * SUB_D;
    float dist = 0.f;
#pragma unroll
    for (int d = 0; d < SUB_D; ++d) { float df = zl[d] - c[d]; dist += df * df; }
    if (dist < bestd) { bestd = dist; best = s; }
  }
  const float* c = cb + ((size_t)k * S_CB + best) * SUB_D;
#pragma unroll
  for (int d = 0; d < SUB_D; ++d)
    zq[(size_t)row * CD_DIM + k * SUB_D + d] = (_Float16)c[d];
  idx_out[(size_t)row * K_CB + k] = (float)best;
  atomicAdd(&hist[k * S_CB + best], 1);
  atomicAdd(sqerr, bestd);
}

__global__ void zero_stats(int* __restrict__ hist, float* __restrict__ sqerr)
{
  int i = blockIdx.x * blockDim.x + threadIdx.x;
  if (i < K_CB * S_CB) hist[i] = 0;
  if (i == 0) sqerr[0] = 0.f;
}

__global__ void finalize_kernel(const float* __restrict__ sqerr,
                                const int* __restrict__ hist,
                                float* __restrict__ out3)
{
  if (threadIdx.x != 0 || blockIdx.x != 0) return;
  float loss = sqerr[0] / (float)(K_CB * B_DIM * L_DIM * SUB_D);
  float perp = 0.f;
  for (int k = 0; k < K_CB; ++k) {
    float Hh = 0.f;
    for (int s = 0; s < S_CB; ++s) {
      float avg = (float)hist[k * S_CB + s] / (float)(B_DIM * L_DIM);
      Hh += avg * logf(fmaxf(avg, 1e-8f));
    }
    perp += expf(-Hh);
  }
  perp *= (1.0f / K_CB);
  out3[0] = loss;
  out3[1] = loss;
  out3[2] = perp;
}

// ---------------------------------------------------------------------------
// Host orchestration
// ---------------------------------------------------------------------------
extern "C" void kernel_launch(void* const* d_in, const int* in_sizes, int n_in,
                              void* d_out, int out_size, void* d_ws, size_t ws_size,
                              hipStream_t stream)
{
  (void)in_sizes; (void)n_in; (void)out_size; (void)ws_size;
  const float* pose6d = (const float*)d_in[0];
  const float* trans  = (const float*)d_in[1];
  const float* obj    = (const float*)d_in[2];
  auto Pp = [&](int i) { return (const float*)d_in[i]; };
  // param leaves (dict insertion order): 3 pe1.w 4 pe1.b 5 pe2.w 6 pe2.b
  // 7 ff1.w 8 ff1.b 9 ff2.w 10 ff2.b 11 enc_in.w 12 enc_in.b
  // 13..24 enc_l0, 25..36 enc_l1, 37 latent.w 38 latent.b 39 codebooks
  // 40 dec_in.w 41 dec_in.b 42 decd_in.w 43 decd_in.b
  // 44..55 dec_l0, 56..67 dec_l1, 68 pose_head.w 69 pose_head.b
  // 70 trans_head.w 71 trans_head.b 72 ph1.w 73 ph1.b 74 ph2.w 75 ph2.b
  // layer offsets: +0 qkv.w +1 qkv.b +2 out.w +3 out.b +4 ff1.w +5 ff1.b
  //                +6 ff2.w +7 ff2.b +8 ln1w +9 ln1b +10 ln2w +11 ln2b

  _Float16* Wk = (_Float16*)d_ws;
  size_t off = 0;
  auto carve = [&](size_t n_halves) { _Float16* p = Wk + off; off += n_halves; return p; };

  _Float16* Xa   = carve((size_t)16384 * 256);
  _Float16* Xt   = carve((size_t)16384 * 256);
  _Float16* Xc   = carve((size_t)16384 * 256);
  _Float16* Qkv  = carve((size_t)16384 * 768);
  _Float16* Big  = carve((size_t)16384 * 1024);     // also frame (16384x416)
  _Float16* Obj16= carve((size_t)8192 * 32);
  _Float16* Zq   = carve((size_t)8192 * 128);
  float*    Ze   = (float*)carve((size_t)8192 * 128 * 2);
  // converted f16 weights
  _Float16* pe1w   = carve(256 * 32);
  _Float16* pe2w   = carve(256 * 256);
  _Float16* ff1w   = carve(256 * FRAME_KP);
  _Float16* ff2w   = carve(256 * 256);
  _Float16* encinw = carve(256 * 256);
  _Float16* Lw[4][4];   // per layer: qkv.w, out.w, ff1.w, ff2.w
  for (int l = 0; l < 4; ++l) {
    Lw[l][0] = carve(768 * 256);
    Lw[l][1] = carve(256 * 256);
    Lw[l][2] = carve(1024 * 256);
    Lw[l][3] = carve(256 * 1024);
  }
  _Float16* latw   = carve(128 * 256);
  _Float16* decinw = carve(256 * 128);
  _Float16* decdw  = carve(256 * 256);
  _Float16* posew  = carve(144 * 256);
  _Float16* transw = carve(16 * 256);   // 3 rows used, padded carve
  _Float16* ph1w   = carve(256 * 128);
  _Float16* ph2w   = carve(768 * 256);
  // small scratch
  float*    ptmean = (float*)carve(8192 * 2);       // 32*256 f32
  _Float16* zqm    = carve(32 * 128);
  _Float16* p1buf  = carve(32 * 256);
  int*      hist   = (int*)carve(512 * 2);
  float*    sqerr  = (float*)carve(2);

  float* outf = (float*)d_out;

  auto cvt = [&](const float* src, _Float16* dst, int n) {
    cvt_f32_f16<<<dim3((n + 255) / 256), dim3(256), 0, stream>>>(src, dst, n);
  };
  auto cvtpad = [&](const float* src, _Float16* dst, int rows, int K, int Kp) {
    cvt_pad_f32_f16<<<dim3((rows * Kp + 255) / 256), dim3(256), 0, stream>>>(
        src, dst, rows, K, Kp);
  };
  auto gemm = [&](const _Float16* A, const _Float16* Wt, const float* bias,
                  void* C, int M, int N, int K, int act, int cf32) {
    dim3 grid((M + 127) / 128, (N + 63) / 64);
    gemm_f16<<<grid, dim3(256), 0, stream>>>(A, Wt, bias, C, M, N, K, act, cf32);
  };

  // ---- one-time per launch: weight conversion (all f16, K padded) ----
  cvtpad(Pp(3), pe1w, 256, 3, 32);
  cvt(Pp(5), pe2w, 256 * 256);
  cvtpad(Pp(7), ff1w, 256, FRAME_K, FRAME_KP);
  cvt(Pp(9), ff2w, 256 * 256);
  cvt(Pp(11), encinw, 256 * 256);
  const int lbase[4] = {13, 25, 44, 56};
  for (int l = 0; l < 4; ++l) {
    cvt(Pp(lbase[l] + 0), Lw[l][0], 768 * 256);
    cvt(Pp(lbase[l] + 2), Lw[l][1], 256 * 256);
    cvt(Pp(lbase[l] + 4), Lw[l][2], 1024 * 256);
    cvt(Pp(lbase[l] + 6), Lw[l][3], 256 * 1024);
  }
  cvt(Pp(37), latw, 128 * 256);
  cvt(Pp(40), decinw, 256 * 128);
  cvt(Pp(42), decdw, 256 * 256);
  cvt(Pp(68), posew, 144 * 256);
  cvt(Pp(70), transw, 3 * 256);
  cvt(Pp(72), ph1w, 256 * 128);
  cvt(Pp(74), ph2w, 768 * 256);
  cvtpad(obj, Obj16, 8192, 3, 32);

  auto layer = [&](int l) {
    int pb = lbase[l];
    gemm(Xa, Lw[l][0], Pp(pb + 1), Qkv, 16384, 768, 256, 0, 0);
    attn_flash<<<dim3(B_DIM * NHEADS * (T_DIM / 16)), dim3(32), 0, stream>>>(
        Qkv, Xc, T_DIM);
    gemm(Xc, Lw[l][1], Pp(pb + 3), Xt, 16384, 256, 256, 0, 0);
    ln_residual<<<dim3(2048), dim3(256), 0, stream>>>(Xa, Xt, Pp(pb + 8),
                                                      Pp(pb + 9), Xc, 16384);
    gemm(Xc, Lw[l][2], Pp(pb + 5), Big, 16384, 1024, 256, 1, 0);
    gemm(Big, Lw[l][3], Pp(pb + 7), Xt, 16384, 256, 1024, 0, 0);
    ln_residual<<<dim3(2048), dim3(256), 0, stream>>>(Xc, Xt, Pp(pb + 10),
                                                      Pp(pb + 11), Xa, 16384);
  };

  // --- point encoder ---
  gemm(Obj16, pe1w, Pp(4), Xt, 8192, 256, 32, 1, 0);
  gemm(Xt, pe2w, Pp(6), Xc, 8192, 256, 256, 1, 0);
  mean_h2f<<<dim3((B_DIM * H_DIM + 255) / 256), dim3(256), 0, stream>>>(
      Xc, ptmean, B_DIM, P_DIM, H_DIM);

  // --- frame MLP + enc_in ---
  int ftot = 16384 * FRAME_KP;
  concat_frame<<<dim3((ftot + 255) / 256), dim3(256), 0, stream>>>(
      pose6d, trans, ptmean, Big, ftot);
  gemm(Big, ff1w, Pp(8), Xt, 16384, 256, FRAME_KP, 1, 0);
  gemm(Xt, ff2w, Pp(10), Xc, 16384, 256, 256, 1, 0);
  gemm(Xc, encinw, Pp(12), Xa, 16384, 256, 256, 0, 0);

  // --- encoder ---
  layer(0);
  layer(1);

  // --- latent + quantize ---
  take_stride2<<<dim3((8192 * 256 + 255) / 256), dim3(256), 0, stream>>>(
      Xa, Xt, 8192 * 256);
  gemm(Xt, latw, Pp(38), Ze, 8192, 128, 256, 0, 1);
  zero_stats<<<dim3(2), dim3(256), 0, stream>>>(hist, sqerr);
  quantize_kernel<<<dim3((32768 + 255) / 256), dim3(256), 0, stream>>>(
      Ze, Pp(39), Zq, outf + IDX_OFF, hist, sqerr);

  // --- decoder ---
  gemm(Zq, decinw, Pp(41), Xt, 8192, 256, 128, 0, 0);
  repeat2<<<dim3((16384 * 256 + 255) / 256), dim3(256), 0, stream>>>(
      Xt, Xc, 16384 * 256);
  gemm(Xc, decdw, Pp(43), Xa, 16384, 256, 256, 0, 0);
  layer(2);
  layer(3);

  // --- heads ---
  gemm(Xa, posew, Pp(69), outf, 16384, 144, 256, 0, 1);
  gemm(Xa, transw, Pp(71), outf + TRANS_OFF, 16384, 3, 256, 0, 1);
  mean_h2h<<<dim3((B_DIM * CD_DIM + 255) / 256), dim3(256), 0, stream>>>(
      Zq, zqm, B_DIM, L_DIM, CD_DIM);
  gemm(zqm, ph1w, Pp(73), p1buf, 32, 256, 128, 1, 0);
  gemm(p1buf, ph2w, Pp(75), outf + PTS_OFF, 32, 768, 256, 0, 1);
  finalize_kernel<<<dim3(1), dim3(1), 0, stream>>>(sqerr, hist, outf + SC_OFF);
}
